// LuongAttnDecoderRNN_79474074845199
// MI455X (gfx1250) — matrix operand order repaired and checked by reading, not compiled, not run on hardware
//
#include <hip/hip_runtime.h>
#include <hip/hip_bf16.h>
#include <math.h>

// Problem constants (from the reference)
#define BS    256
#define NN    51200
#define HDIM  512
#define OUTF  256

typedef __attribute__((ext_vector_type(16))) __bf16       v16bf;
typedef __attribute__((ext_vector_type(8)))  float        v8f;
typedef __attribute__((ext_vector_type(8)))  unsigned int v8u;
typedef __attribute__((ext_vector_type(4)))  unsigned int v4u;
typedef __attribute__((ext_vector_type(8)))  int          v8i;
typedef __attribute__((ext_vector_type(4)))  int          v4i;

union BF16x16 { v16bf v; unsigned short s[16]; v8u u; };

__device__ inline unsigned short f2bf(float f){
  union { float f; unsigned u; } x; x.f = f;
  unsigned u = x.u;
  u += 0x7FFFu + ((u >> 16) & 1u);   // round-to-nearest-even
  return (unsigned short)(u >> 16);
}

__device__ inline v8f wmma_bf16(v16bf a, v16bf b, v8f c){
  // D = A(16x32 bf16) x B(32x16 bf16) + C(16x16 f32)
  return __builtin_amdgcn_wmma_f32_16x16x32_bf16(false, a, false, b, (short)0, c, false, false);
}

// A-fragment (16x32, bf16) from row-major f32 matrix with row stride 512.
// Lane layout: lanes 0-15 -> M=lane, K = k..k+15;
// lanes 16-31 -> M=lane-16, K = k+16..k+31.
__device__ inline v16bf load_a_f32(const float* A, int row0, int k, int relu){
  int lane = threadIdx.x & 31;
  const float* p = A + (long long)(row0 + (lane & 15)) * 512 + k + ((lane >> 4) * 16);
  BF16x16 t;
  #pragma unroll
  for (int i = 0; i < 8; ++i){
    float a = p[i];      if (relu) a = fmaxf(a, 0.f);
    float b = p[i + 8]; if (relu) b = fmaxf(b, 0.f);
    t.s[i] = f2bf(a); t.s[8 + i] = f2bf(b);
  }
  return t.v;
}

// B-fragment (32x16, bf16) of W^T taken from row-major W [N x K]:
// B[k][n] = W[n][k], so lane owning column n reads W row n contiguously.
// lanes 0-15 -> N=lane, K=k..k+15 ; lanes 16-31 -> N=lane-16, K=k+16..k+31.
__device__ inline v16bf load_b_wt(const float* W, int col0, int k, int ldw){
  int lane = threadIdx.x & 31;
  const float* p = W + (long long)(col0 + (lane & 15)) * ldw + k + ((lane >> 4) * 16);
  BF16x16 t;
  #pragma unroll
  for (int i = 0; i < 16; ++i) t.s[i] = f2bf(p[i]);
  return t.v;
}

// ---------------------------------------------------------------------------
// Generic linear: C[M,N] = act( act_in(A) @ W^T + bias ).  A is split in up to
// three 512-wide column chunks (for the concat in `compress`).  One wave per
// 16x16 C tile, 8 waves per block.
// ---------------------------------------------------------------------------
__global__ __launch_bounds__(256) void k_linear(
    const float* A0, const float* A1, const float* A2,
    const float* __restrict__ W, const float* __restrict__ bias,
    float* __restrict__ C, int tiles_n, int N, int K,
    int relu_in, int relu_out, unsigned short* bf_copy)
{
  int wid  = threadIdx.x >> 5;
  int lane = threadIdx.x & 31;
  int tile = blockIdx.x * 8 + wid;
  int row0 = (tile / tiles_n) * 16;
  int col0 = (tile % tiles_n) * 16;
  v8f acc = {};
  int nk = K >> 5;
  for (int kk = 0; kk < nk; ++kk){
    int kg = kk << 5;
    const float* A = A0; int kl = kg;
    if      (kg >= 1024){ A = A2; kl = kg - 1024; }
    else if (kg >= 512) { A = A1; kl = kg - 512; }
    v16bf a = load_a_f32(A, row0, kl, relu_in);
    v16bf b = load_b_wt(W, col0, kg, K);
    acc = wmma_bf16(a, b, acc);
  }
  int c  = col0 + (lane & 15);
  int mb = row0 + ((lane >> 4) * 8);
  float bv = bias[c];
  #pragma unroll
  for (int r = 0; r < 8; ++r){
    float v = acc[r] + bv;
    if (relu_out) v = fmaxf(v, 0.f);
    C[(long long)(mb + r) * N + c] = v;
    if (bf_copy) bf_copy[(long long)(mb + r) * N + c] = f2bf(v);
  }
}

// ---------------------------------------------------------------------------
// Fused GRU step: 6 WMMA accumulators per tile (r/z/n gates x {ih,hh}),
// then gate math + new hidden written straight to d_out's hidden segment.
// ---------------------------------------------------------------------------
__global__ __launch_bounds__(256) void k_gru(
  const float* __restrict__ x, const float* __restrict__ hprev,
  const float* __restrict__ W_ih, const float* __restrict__ W_hh,
  const float* __restrict__ b_ih, const float* __restrict__ b_hh,
  float* __restrict__ hout)
{
  int wid = threadIdx.x >> 5, lane = threadIdx.x & 31;
  int tile = blockIdx.x * 8 + wid;
  int row0 = (tile >> 5) * 16;   // tiles_n = 512/16 = 32
  int col0 = (tile & 31) * 16;
  v8f ir = {}, iz = {}, inn = {}, hr = {}, hz = {}, hn = {};
  for (int k = 0; k < 512; k += 32){
    v16bf ax = load_a_f32(x,     row0, k, 0);
    v16bf ah = load_a_f32(hprev, row0, k, 0);
    ir  = wmma_bf16(ax, load_b_wt(W_ih, col0,        k, 512), ir);
    iz  = wmma_bf16(ax, load_b_wt(W_ih, col0 + 512,  k, 512), iz);
    inn = wmma_bf16(ax, load_b_wt(W_ih, col0 + 1024, k, 512), inn);
    hr  = wmma_bf16(ah, load_b_wt(W_hh, col0,        k, 512), hr);
    hz  = wmma_bf16(ah, load_b_wt(W_hh, col0 + 512,  k, 512), hz);
    hn  = wmma_bf16(ah, load_b_wt(W_hh, col0 + 1024, k, 512), hn);
  }
  int c  = col0 + (lane & 15);
  int mb = row0 + ((lane >> 4) * 8);
  float bir = b_ih[c], biz = b_ih[512 + c], bin = b_ih[1024 + c];
  float bhr = b_hh[c], bhz = b_hh[512 + c], bhn = b_hh[1024 + c];
  #pragma unroll
  for (int r = 0; r < 8; ++r){
    int row = mb + r;
    float gr = 1.f / (1.f + __expf(-((ir[r]  + bir) + (hr[r] + bhr))));
    float gz = 1.f / (1.f + __expf(-((iz[r]  + biz) + (hz[r] + bhz))));
    float gn = tanhf((inn[r] + bin) + gr * (hn[r] + bhn));
    float hp = hprev[(long long)row * 512 + c];
    hout[(long long)row * 512 + c] = (1.f - gz) * gn + gz * hp;
  }
}

// ---------------------------------------------------------------------------
// Attention scores: scores[n][b] = <nodes[n], rnn_out[b]>.
// Block = 128 nodes x all 256 columns.  B (bf16 rnn_out, 256 KB) is staged
// once into LDS with the Tensor Data Mover; K-loop streams f32 nodes and
// converts to bf16 for WMMA.  Epilogue transposes 16x16 tiles through LDS so
// stores to the scores scratch are 64B contiguous per lane.
// ---------------------------------------------------------------------------
#define SMEM_B_BYTES (256 * 512 * 2)

__global__ __launch_bounds__(256) void k_scores(
  const float* __restrict__ nodes, const unsigned short* __restrict__ rnnbf,
  float* __restrict__ scores)
{
  extern __shared__ unsigned char smem[];
  unsigned short* sB = (unsigned short*)smem;                 // [256][512] bf16
  float*          sT = (float*)(smem + SMEM_B_BYTES);         // 8 waves x 16x16 f32
  int tid = threadIdx.x, wid = tid >> 5, lane = tid & 31;

#if defined(__HIP_DEVICE_COMPILE__) && __has_builtin(__builtin_amdgcn_tensor_load_to_lds)
  if (wid == 0){
    unsigned lds = (unsigned)(unsigned long long)(void*)sB;   // LDS byte offset
    unsigned long long ga = (unsigned long long)rnnbf;
    // D# group0: count=1 | lds_addr | global_addr | type=2
    v4u g0 = { 1u, lds, (unsigned)ga,
               (unsigned)((ga >> 32) & 0x01FFFFFFull) | 0x80000000u };
    // D# group1: data_size=8B, 1D tile: tensor_dim0 = tile_dim0 = 32768 elems
    // (= 256 KB), tensor_dim0_stride = 32768, no multicast/pad/iterate.
    v8i g1 = { (int)(3u << 16),          // data_size=8B
               (int)(0x8000u << 16),     // tensor_dim0[15:0]
               (int)(1u << 16),          // tensor_dim0[31:16]=0, tensor_dim1=1
               (int)0x80000000u,         // tile_dim0 = 0x8000
               0,                        // tile_dim1/2 unused
               32768,                    // tensor_dim0_stride[31:0]
               0, 0 };
    v4i g2 = {0,0,0,0};
    v4i g3 = {0,0,0,0};
    v8i g4 = {0,0,0,0,0,0,0,0};
    __builtin_amdgcn_tensor_load_to_lds(g0, g1, g2, g3, g4, 0);
  #if __has_builtin(__builtin_amdgcn_s_wait_tensorcnt)
    __builtin_amdgcn_s_wait_tensorcnt(0);
  #else
    asm volatile("s_wait_tensorcnt 0x0" ::: "memory");
  #endif
  }
#else
  { // cooperative fallback copy (also the host-pass parse path)
    const v4u* src = (const v4u*)rnnbf;
    v4u* dst = (v4u*)sB;
    for (int i = tid; i < SMEM_B_BYTES / 16; i += 256) dst[i] = src[i];
  }
#endif
  __syncthreads();

  int n0 = blockIdx.x * 128 + wid * 16;   // this wave's 16 node rows
  v8f zero = {};
  v8f acc[16];
  #pragma unroll
  for (int t = 0; t < 16; ++t) acc[t] = zero;

  for (int k = 0; k < 512; k += 32){
    __builtin_prefetch(nodes + (long long)(n0 + (lane & 15)) * 512 + k + 64, 0, 0);
    v16bf a = load_a_f32(nodes, n0, k, 0);
    #pragma unroll
    for (int nt = 0; nt < 16; ++nt){
      int b = nt * 16 + (lane & 15);
      const v8u* p = (const v8u*)(sB + b * 512 + k + ((lane >> 4) << 4));
      BF16x16 t; t.u = *p;
      acc[nt] = wmma_bf16(a, t.v, acc[nt]);
    }
  }

  // transpose each 16x16 tile through per-wave LDS scratch, store row-major
  float* tw = sT + wid * 256;
  int cc = lane & 15, mbl = (lane >> 4) * 8;
  for (int nt = 0; nt < 16; ++nt){
    #pragma unroll
    for (int r = 0; r < 8; ++r) tw[(mbl + r) * 16 + cc] = acc[nt][r];
    asm volatile("s_wait_dscnt 0x0" ::: "memory");
    if (lane < 16){
      float4* d = (float4*)(scores + (long long)(n0 + lane) * 256 + nt * 16);
      const float4* s4 = (const float4*)(tw + lane * 16);
      d[0] = s4[0]; d[1] = s4[1]; d[2] = s4[2]; d[3] = s4[3];
    }
    asm volatile("s_wait_dscnt 0x0" ::: "memory");
  }
}

// Segment boundaries via binary search (batch is sorted, 256 segments).
__global__ void k_segbounds(const int* __restrict__ batch, int* __restrict__ bnd){
  int s = threadIdx.x;
  int lo = 0, hi = NN;
  while (lo < hi){ int mid = (lo + hi) >> 1; if (batch[mid] < s) lo = mid + 1; else hi = mid; }
  int st = lo; hi = NN;
  while (lo < hi){ int mid = (lo + hi) >> 1; if (batch[mid] < s + 1) lo = mid + 1; else hi = mid; }
  bnd[s] = st; bnd[256 + s] = lo;
}

// Per-(segment, column) max and exp-sum; one block per segment, one column
// per thread, reads of scores are coalesced across threads.
__global__ __launch_bounds__(256) void k_stats(
  const float* __restrict__ scores, const int* __restrict__ bnd,
  float* __restrict__ segmax, float* __restrict__ segsum)
{
  int s = blockIdx.x, b = threadIdx.x;
  int st = bnd[s], en = bnd[256 + s];
  float mx = -3.4e38f;
  for (int n = st; n < en; ++n) mx = fmaxf(mx, scores[(long long)n * 256 + b]);
  float sum = 0.f;
  for (int n = st; n < en; ++n) sum += __expf(scores[(long long)n * 256 + b] - mx);
  segmax[s * 256 + b] = mx;
  segsum[s * 256 + b] = (en > st) ? sum : 1.f;
}

// attn_weights[b][n]: coalesced over n (thread = node), loop over b.
__global__ __launch_bounds__(128) void k_final(
  const float* __restrict__ scores, const int* __restrict__ batch,
  const float* __restrict__ segmax, const float* __restrict__ segsum,
  float* __restrict__ att)
{
  int n = blockIdx.x * 128 + threadIdx.x;
  int s = batch[n];
  const float* sm = segmax + s * 256;
  const float* ss = segsum + s * 256;
  const float* sr = scores + (long long)n * 256;
  for (int b = 0; b < 256; ++b){
    float w = __expf(sr[b] - sm[b]) / ss[b];
    att[(long long)b * NN + n] = w;
  }
}

// context[s] = sum over segment of w_diag[n] * nodes[n]; one block per
// segment, each thread owns 2 of the 512 feature channels in registers.
__global__ __launch_bounds__(256) void k_context(
  const float* __restrict__ scores, const float* __restrict__ nodes,
  const int* __restrict__ bnd, const float* __restrict__ segmax,
  const float* __restrict__ segsum, float* __restrict__ ctx)
{
  int s = blockIdx.x, t = threadIdx.x;
  int st = bnd[s], en = bnd[256 + s];
  float mx  = segmax[s * 256 + s];
  float inv = 1.f / segsum[s * 256 + s];
  float c0 = 0.f, c1 = 0.f;
  for (int n = st; n < en; ++n){
    float w = __expf(scores[(long long)n * 256 + s] - mx) * inv;
    c0 += w * nodes[(long long)n * 512 + t];
    c1 += w * nodes[(long long)n * 512 + t + 256];
  }
  ctx[s * 512 + t]       = c0;
  ctx[s * 512 + t + 256] = c1;
}

// ---------------------------------------------------------------------------
extern "C" void kernel_launch(void* const* d_in, const int* in_sizes, int n_in,
                              void* d_out, int out_size, void* d_ws, size_t ws_size,
                              hipStream_t stream)
{
  (void)in_sizes; (void)n_in; (void)out_size; (void)ws_size;
  const float* input_seq   = (const float*)d_in[0];
  const float* last_hidden = (const float*)d_in[1];
  const float* nodes       = (const float*)d_in[2];
  const int*   batch       = (const int*)d_in[3];
  const float* enc_w  = (const float*)d_in[4];
  const float* enc_b  = (const float*)d_in[5];
  const float* W_ih   = (const float*)d_in[6];
  const float* W_hh   = (const float*)d_in[7];
  const float* b_ih   = (const float*)d_in[8];
  const float* b_hh   = (const float*)d_in[9];
  const float* proj_w = (const float*)d_in[10];
  const float* proj_b = (const float*)d_in[11];
  const float* comp_w = (const float*)d_in[12];
  const float* comp_b = (const float*)d_in[13];
  const float* out_w  = (const float*)d_in[14];
  const float* out_b  = (const float*)d_in[15];

  float* out = (float*)d_out;               // [256,256]
  float* hid = out + 65536;                 // [1,256,512]
  float* att = out + 65536 + 131072;        // [256,51200]

  float* ws          = (float*)d_ws;
  float* ws_x        = ws;                                   // [256,512]
  float* ws_rnn      = ws + 131072;                          // [256,512]
  unsigned short* ws_rnnbf = (unsigned short*)(ws + 262144); // [256,512] bf16
  float* ws_scores   = ws + 327680;                          // [51200,256]
  float* ws_segmax   = ws + 327680 + 13107200;               // [256,256]
  float* ws_segsum   = ws_segmax + 65536;                    // [256,256]
  int*   ws_bnd      = (int*)(ws_segsum + 65536);            // [512] ints
  float* ws_ctx      = (float*)(ws_bnd + 512);               // [256,512]
  float* ws_cc       = ws_ctx + 131072;                      // [256,512]

  // 1) x = relu(input @ enc_w^T + enc_b)
  k_linear<<<64, 256, 0, stream>>>(input_seq, nullptr, nullptr, enc_w, enc_b,
                                   ws_x, 32, 512, 512, 0, 1, nullptr);
  // 2) h = GRU(x, last_hidden)  -> d_out hidden segment
  k_gru<<<64, 256, 0, stream>>>(ws_x, last_hidden, W_ih, W_hh, b_ih, b_hh, hid);
  // 3) rnn_out = relu(h) @ proj_w^T + proj_b   (+ bf16 copy for attention B)
  k_linear<<<64, 256, 0, stream>>>(hid, nullptr, nullptr, proj_w, proj_b,
                                   ws_rnn, 32, 512, 512, 1, 0, ws_rnnbf);
  // 4) attention
  k_segbounds<<<1, 256, 0, stream>>>(batch, ws_bnd);
  k_scores<<<400, 256, SMEM_B_BYTES + 8 * 256 * 4, stream>>>(nodes, ws_rnnbf, ws_scores);
  k_stats<<<256, 256, 0, stream>>>(ws_scores, ws_bnd, ws_segmax, ws_segsum);
  k_final<<<400, 128, 0, stream>>>(ws_scores, batch, ws_segmax, ws_segsum, att);
  k_context<<<256, 256, 0, stream>>>(ws_scores, nodes, ws_bnd, ws_segmax, ws_segsum, ws_ctx);
  // 5) concat_out = relu([rnn_out|context|x] @ comp_w^T + comp_b)
  k_linear<<<64, 256, 0, stream>>>(ws_rnn, ws_ctx, ws_x, comp_w, comp_b,
                                   ws_cc, 32, 512, 1536, 0, 1, nullptr);
  // 6) output = concat_out @ out_w^T + out_b  -> d_out
  k_linear<<<32, 256, 0, stream>>>(ws_cc, nullptr, nullptr, out_w, out_b,
                                   out, 16, 256, 512, 0, 0, nullptr);
}
